// SUVMLoSSNet_84215718740567
// MI455X (gfx1250) — matrix operand order repaired and checked by reading, not compiled, not run on hardware
//
#include <hip/hip_runtime.h>
#include <math.h>

typedef __attribute__((ext_vector_type(16))) _Float16 v16h;
typedef __attribute__((ext_vector_type(8)))  float    v8f;

#define LQ     4096      // H*W
#define CCH    96        // channels
#define DIN    144
#define DPAD   160       // DIN padded to multiple of 32
#define SST    16        // state size
#define RR     6
#define NDIR   4
#define MROWS  (NDIR*LQ) // 16384
#define NPROJ  288       // 2*DIN
#define NDBL   48        // R+2S=38 padded to 48
#define NCC    32        // CAB hidden channels
#define NCHK   64        // scan chunks
#define CHKL   (LQ/NCHK) // 64 steps per chunk

__device__ __forceinline__ float siluf(float x){ return x / (1.f + expf(-x)); }
__device__ __forceinline__ float softplusf(float x){ return (x > 20.f) ? x : log1pf(expf(x)); }

// ---------------- LN1 + scatter into 4 scan orders (f16 for WMMA A) ----------
__global__ void k_ln1_gather(const float* __restrict__ x,
                             const float* __restrict__ g, const float* __restrict__ b,
                             const int* __restrict__ iids,          // [4][L]
                             float* __restrict__ xn, _Float16* __restrict__ xs16)
{
  int l = blockIdx.x * blockDim.x + threadIdx.x;
  if (l >= LQ) return;
  const float* row = x + (size_t)l * CCH;
  float m = 0.f;
  for (int c = 0; c < CCH; ++c) m += row[c];
  m *= (1.f / CCH);
  float v = 0.f;
  for (int c = 0; c < CCH; ++c) { float d = row[c] - m; v += d * d; }
  v *= (1.f / CCH);
  float rs = rsqrtf(v + 1e-5f);
  int j0 = iids[0*LQ + l], j1 = iids[1*LQ + l], j2 = iids[2*LQ + l], j3 = iids[3*LQ + l];
  for (int c = 0; c < CCH; ++c) {
    float yv = (row[c] - m) * rs * g[c] + b[c];
    xn[(size_t)l * CCH + c] = yv;
    _Float16 hv = (_Float16)yv;
    xs16[(size_t)(0*LQ + j0) * CCH + c] = hv;
    xs16[(size_t)(1*LQ + j1) * CCH + c] = hv;
    xs16[(size_t)(2*LQ + j2) * CCH + c] = hv;
    xs16[(size_t)(3*LQ + j3) * CCH + c] = hv;
  }
}

// ---------------- weight f32 -> f16 with zero padding -------------------------
__global__ void k_wcvt(const float* __restrict__ src, _Float16* __restrict__ dst,
                       int NrowsDst, int NrowsReal, int Kdst, int Kreal)
{
  int idx = blockIdx.x * blockDim.x + threadIdx.x;
  if (idx >= NrowsDst * Kdst) return;
  int n = idx / Kdst, k = idx % Kdst;
  dst[idx] = (n < NrowsReal && k < Kreal) ? (_Float16)src[(size_t)n * Kreal + k]
                                          : (_Float16)0.f;
}

// ---------------- WMMA GEMM: C[M,N](f32) = A[M,K](f16) * W[N,K](f16)^T -------
// one wave per 16x16 tile; K multiple of 32; M,N multiples of 16
__global__ void k_gemm_wmma(const _Float16* __restrict__ A, const _Float16* __restrict__ Bw,
                            float* __restrict__ Cout, int Mtiles, int Ntiles,
                            int lda, int ldb, int ldc, int K)
{
  int wave = (int)((blockIdx.x * blockDim.x + threadIdx.x) >> 5);
  int lane = threadIdx.x & 31;
  if (wave >= Mtiles * Ntiles) return;
  int tm = wave / Ntiles, tn = wave % Ntiles;
  int r0 = lane & 15, half = lane >> 4;
  const _Float16* arow = A  + (size_t)(tm * 16 + r0) * lda;   // row m = r0
  const _Float16* brow = Bw + (size_t)(tn * 16 + r0) * ldb;   // col n = r0 (W is [N][K])
  v8f acc = {};
  for (int k0 = 0; k0 < K; k0 += 32) {
    v16h a, b;
#pragma unroll
    for (int j = 0; j < 8; ++j) {
      a[j]     = arow[k0 + 8 * half + j];          // K = 8*half + 0..7
      a[j + 8] = arow[k0 + 16 + 8 * half + j];     // K = 16 + 8*half + 0..7
    }
#pragma unroll
    for (int j = 0; j < 8; ++j) {
      b[j]     = brow[k0 + 8 * half + j];          // K = 8*half + 0..7
      b[j + 8] = brow[k0 + 16 + 8 * half + j];     // K = 16 + 8*half + 0..7
    }
    acc = __builtin_amdgcn_wmma_f32_16x16x32_f16(false, a, false, b,
                                                 (short)0, acc, false, false);
  }
#pragma unroll
  for (int r = 0; r < 8; ++r)
    Cout[(size_t)(tm * 16 + r + 8 * half) * ldc + tn * 16 + r0] = acc[r];
}

// ---------------- depthwise causal conv3 + SiLU ------------------------------
__global__ void k_conv_silu(const float* __restrict__ xz,
                            const float* __restrict__ cw, const float* __restrict__ cb,
                            float* __restrict__ xm, _Float16* __restrict__ xm16)
{
  int idx = blockIdx.x * blockDim.x + threadIdx.x;
  if (idx >= MROWS * DPAD) return;
  int row = idx / DPAD, d = idx % DPAD;
  if (d >= DIN) { xm16[idx] = (_Float16)0.f; return; }
  int l = row % LQ;
  const float* xzr = xz + (size_t)row * NPROJ;
  float v2 = xzr[d];
  float v1 = (l >= 1) ? xzr[d - NPROJ]     : 0.f;
  float v0 = (l >= 2) ? xzr[d - 2 * NPROJ] : 0.f;
  float xc = cw[d * 3 + 0] * v0 + cw[d * 3 + 1] * v1 + cw[d * 3 + 2] * v2 + cb[d];
  float sm = siluf(xc);
  xm[(size_t)row * DIN + d] = sm;
  xm16[idx] = (_Float16)sm;
}

// ---------------- dt = softplus(dbl[:, :6] @ dtw^T + dtb) --------------------
__global__ void k_dt(const float* __restrict__ dbl, const float* __restrict__ dtw,
                     const float* __restrict__ dtb, float* __restrict__ dt)
{
  int idx = blockIdx.x * blockDim.x + threadIdx.x;
  if (idx >= MROWS * DIN) return;
  int row = idx / DIN, d = idx % DIN;
  const float* dr = dbl + (size_t)row * NDBL;
  float acc = dtb[d];
#pragma unroll
  for (int r = 0; r < RR; ++r) acc += dr[r] * dtw[d * RR + r];
  dt[idx] = softplusf(acc);
}

// ---------------- chunked scan, phase 1: per-chunk aggregates ----------------
// halfwave = (dir*DIN+d)*NCHK + chunk ; lane = state s
// aggregate: A_c = exp(A * sum(dt)), B_c = chunk-local scan end state (h_in = 0)
__global__ void k_scan_p1(const float* __restrict__ dt, const float* __restrict__ xm,
                          const float* __restrict__ dbl, const float* __restrict__ alog,
                          float* __restrict__ aP, float* __restrict__ bAgg)
{
  int hw = blockIdx.x * (blockDim.x >> 4) + (threadIdx.x >> 4);
  int s  = threadIdx.x & 15;
  if (hw >= NDIR * DIN * NCHK) return;
  int chunk = hw % NCHK;
  int dd    = hw / NCHK;           // dir*DIN + d
  int d = dd % DIN, dir = dd / DIN;
  float Aval = -expf(alog[d * SST + s]);
  float h = 0.f, sdt = 0.f;
  size_t row0 = (size_t)dir * LQ + (size_t)chunk * CHKL;
  for (int l = 0; l < CHKL; ++l) {
    size_t row = row0 + l;
    float dtv = dt[row * DIN + d];
    float xv  = xm[row * DIN + d];
    float Bv  = dbl[row * NDBL + RR + s];
    float dA  = expf(dtv * Aval);
    sdt += dtv;
    h = dA * h + dtv * xv * Bv;
  }
  size_t o = ((size_t)dd * NCHK + chunk) * SST + s;
  aP[o]   = expf(sdt * Aval);
  bAgg[o] = h;
}

// ---------------- chunked scan, phase 2: prefix over chunk aggregates --------
// thread per (dir,d,s); writes exclusive entry state h_in per chunk
__global__ void k_scan_p2(const float* __restrict__ aP, const float* __restrict__ bAgg,
                          float* __restrict__ hin)
{
  int idx = blockIdx.x * blockDim.x + threadIdx.x;
  if (idx >= NDIR * DIN * SST) return;
  int s = idx % SST, dd = idx / SST;
  float h = 0.f;
  for (int c = 0; c < NCHK; ++c) {
    size_t o = ((size_t)dd * NCHK + c) * SST + s;
    hin[o] = h;
    h = aP[o] * h + bAgg[o];
  }
}

// ---------------- chunked scan, phase 3: within-chunk scan + y ---------------
__global__ void k_scan_p3(const float* __restrict__ dt, const float* __restrict__ xm,
                          const float* __restrict__ dbl, const float* __restrict__ alog,
                          const float* __restrict__ Dp, const float* __restrict__ hin,
                          float* __restrict__ y)
{
  int hw = blockIdx.x * (blockDim.x >> 4) + (threadIdx.x >> 4);
  int s  = threadIdx.x & 15;
  if (hw >= NDIR * DIN * NCHK) return;
  int chunk = hw % NCHK;
  int dd    = hw / NCHK;
  int d = dd % DIN, dir = dd / DIN;
  float Aval = -expf(alog[d * SST + s]);
  float dpv  = Dp[d];
  float h = hin[((size_t)dd * NCHK + chunk) * SST + s];
  size_t row0 = (size_t)dir * LQ + (size_t)chunk * CHKL;
  for (int l = 0; l < CHKL; ++l) {
    size_t row = row0 + l;
    float dtv = dt[row * DIN + d];
    float xv  = xm[row * DIN + d];
    float Bv  = dbl[row * NDBL + RR + s];
    float Cv  = dbl[row * NDBL + RR + SST + s];
    float dA  = expf(dtv * Aval);
    h = dA * h + dtv * xv * Bv;
    float contrib = h * Cv;
    contrib += __shfl_xor(contrib, 8, 16);
    contrib += __shfl_xor(contrib, 4, 16);
    contrib += __shfl_xor(contrib, 2, 16);
    contrib += __shfl_xor(contrib, 1, 16);
    if (s == 0) y[row * DIN + d] = contrib + dpv * xv;
  }
}

// ---------------- gate: yg = y * silu(z) -> f16 (padded) ---------------------
__global__ void k_gate(const float* __restrict__ y, const float* __restrict__ xz,
                       _Float16* __restrict__ yg16)
{
  int idx = blockIdx.x * blockDim.x + threadIdx.x;
  if (idx >= MROWS * DPAD) return;
  int row = idx / DPAD, d = idx % DPAD;
  if (d >= DIN) { yg16[idx] = (_Float16)0.f; return; }
  float z = xz[(size_t)row * NPROJ + DIN + d];
  yg16[idx] = (_Float16)(y[(size_t)row * DIN + d] * siluf(z));
}

// ---------------- inverse-gather merge (skip collapses to 4*ss*xn) -----------
__global__ void k_merge(const float* __restrict__ yo, const float* __restrict__ xn,
                        const int* __restrict__ iids, const float* __restrict__ ss,
                        float* __restrict__ x2)
{
  int idx = blockIdx.x * blockDim.x + threadIdx.x;
  if (idx >= LQ * CCH) return;
  int l = idx / CCH, c = idx % CCH;
  float acc = 4.f * ss[c] * xn[idx];
#pragma unroll
  for (int dir = 0; dir < NDIR; ++dir) {
    int j = iids[dir * LQ + l];
    acc += yo[(size_t)(dir * LQ + j) * CCH + c];
  }
  x2[idx] = acc;
}

// ---------------- plain LayerNorm --------------------------------------------
__global__ void k_ln(const float* __restrict__ x, const float* __restrict__ g,
                     const float* __restrict__ b, float* __restrict__ o)
{
  int l = blockIdx.x * blockDim.x + threadIdx.x;
  if (l >= LQ) return;
  const float* row = x + (size_t)l * CCH;
  float m = 0.f;
  for (int c = 0; c < CCH; ++c) m += row[c];
  m *= (1.f / CCH);
  float v = 0.f;
  for (int c = 0; c < CCH; ++c) { float d = row[c] - m; v += d * d; }
  v *= (1.f / CCH);
  float rs = rsqrtf(v + 1e-5f);
  for (int c = 0; c < CCH; ++c)
    o[(size_t)l * CCH + c] = (row[c] - m) * rs * g[c] + b[c];
}

// ---------------- CAB conv1 3x3 96->32 + exact GELU --------------------------
__global__ void k_conv1(const float* __restrict__ xn2, const float* __restrict__ w,
                        const float* __restrict__ bias, float* __restrict__ h1)
{
  int idx = blockIdx.x * blockDim.x + threadIdx.x;
  if (idx >= NCC * LQ) return;
  int oc = idx / LQ, p = idx % LQ;
  int hh = p >> 6, ww = p & 63;
  float acc = bias[oc];
  const float* wk = w + (size_t)oc * CCH * 9;
  for (int kh = 0; kh < 3; ++kh) {
    int y0 = hh + kh - 1;
    if ((unsigned)y0 > 63u) continue;
    for (int kw = 0; kw < 3; ++kw) {
      int x0 = ww + kw - 1;
      if ((unsigned)x0 > 63u) continue;
      const float* px = xn2 + (size_t)(y0 * 64 + x0) * CCH;
#pragma unroll 8
      for (int ic = 0; ic < CCH; ++ic)
        acc += wk[ic * 9 + kh * 3 + kw] * px[ic];
    }
  }
  h1[idx] = 0.5f * acc * (1.f + erff(acc * 0.70710678118654752f));
}

// ---------------- CAB conv2 3x3 32->96 ---------------------------------------
__global__ void k_conv2(const float* __restrict__ h1, const float* __restrict__ w,
                        const float* __restrict__ bias, float* __restrict__ h2)
{
  int idx = blockIdx.x * blockDim.x + threadIdx.x;
  if (idx >= CCH * LQ) return;
  int oc = idx / LQ, p = idx % LQ;
  int hh = p >> 6, ww = p & 63;
  float acc = bias[oc];
  const float* wk = w + (size_t)oc * NCC * 9;
  for (int kh = 0; kh < 3; ++kh) {
    int y0 = hh + kh - 1;
    if ((unsigned)y0 > 63u) continue;
    for (int kw = 0; kw < 3; ++kw) {
      int x0 = ww + kw - 1;
      if ((unsigned)x0 > 63u) continue;
      int pp = y0 * 64 + x0;
#pragma unroll 8
      for (int ic = 0; ic < NCC; ++ic)
        acc += wk[ic * 9 + kh * 3 + kw] * h1[(size_t)ic * LQ + pp];
    }
  }
  h2[idx] = acc;
}

// ---------------- channel attention (pool -> 3 -> 96, sigmoid) ---------------
__global__ void k_attn(const float* __restrict__ h2, const float* __restrict__ a1w,
                       const float* __restrict__ a1b, const float* __restrict__ a2w,
                       const float* __restrict__ a2b, float* __restrict__ attn)
{
  __shared__ float pch[CCH];
  __shared__ float hid[3];
  int c = threadIdx.x;
  if (c < CCH) {
    float sum = 0.f;
    const float* row = h2 + (size_t)c * LQ;
    for (int i = 0; i < LQ; ++i) sum += row[i];
    pch[c] = sum * (1.f / LQ);
  }
  __syncthreads();
  if (c < 3) {
    float a = a1b[c];
    for (int k = 0; k < CCH; ++k) a += a1w[c * CCH + k] * pch[k];
    hid[c] = fmaxf(a, 0.f);
  }
  __syncthreads();
  if (c < CCH) {
    float a = a2b[c];
    for (int q = 0; q < 3; ++q) a += a2w[c * 3 + q] * hid[q];
    attn[c] = 1.f / (1.f + expf(-a));
  }
}

// ---------------- final residual combine -------------------------------------
__global__ void k_combine(const float* __restrict__ x2, const float* __restrict__ h2,
                          const float* __restrict__ attn, const float* __restrict__ ss2,
                          float* __restrict__ out)
{
  int idx = blockIdx.x * blockDim.x + threadIdx.x;
  if (idx >= LQ * CCH) return;
  int l = idx / CCH, c = idx % CCH;
  out[idx] = x2[idx] * ss2[c] + h2[(size_t)c * LQ + l] * attn[c];
}

extern "C" void kernel_launch(void* const* d_in, const int* in_sizes, int n_in,
                              void* d_out, int out_size, void* d_ws, size_t ws_size,
                              hipStream_t stream)
{
  (void)in_sizes; (void)n_in; (void)out_size; (void)ws_size;
  const float* x_in     = (const float*)d_in[0];
  const int*   inv_ids  = (const int*)  d_in[2];
  const float* ln1_g    = (const float*)d_in[3];
  const float* ln1_b    = (const float*)d_in[4];
  const float* in_w     = (const float*)d_in[5];
  const float* conv_w   = (const float*)d_in[6];
  const float* conv_b   = (const float*)d_in[7];
  const float* xp_w     = (const float*)d_in[8];
  const float* dt_w     = (const float*)d_in[9];
  const float* dt_b     = (const float*)d_in[10];
  const float* a_log    = (const float*)d_in[11];
  const float* Dp       = (const float*)d_in[12];
  const float* out_w    = (const float*)d_in[13];
  const float* ss       = (const float*)d_in[14];
  const float* ss2      = (const float*)d_in[15];
  const float* ln2_g    = (const float*)d_in[16];
  const float* ln2_b    = (const float*)d_in[17];
  const float* c1w      = (const float*)d_in[18];
  const float* c1b      = (const float*)d_in[19];
  const float* c2w      = (const float*)d_in[20];
  const float* c2b      = (const float*)d_in[21];
  const float* a1w      = (const float*)d_in[22];
  const float* a1b      = (const float*)d_in[23];
  const float* a2w      = (const float*)d_in[24];
  const float* a2b      = (const float*)d_in[25];

  char* ws = (char*)d_ws;
  size_t off = 0;
  auto alloc = [&](size_t bytes) -> char* {
    char* p = ws + off;
    off = (off + bytes + 255) & ~(size_t)255;
    return p;
  };
  float*     xbuf   = (float*)    alloc((size_t)LQ * CCH * 4);
  float*     xn     = (float*)    alloc((size_t)LQ * CCH * 4);
  _Float16*  xs16   = (_Float16*) alloc((size_t)MROWS * CCH * 2);
  float*     xz     = (float*)    alloc((size_t)MROWS * NPROJ * 4);
  float*     xm     = (float*)    alloc((size_t)MROWS * DIN * 4);
  _Float16*  xm16   = (_Float16*) alloc((size_t)MROWS * DPAD * 2);
  float*     dbl    = (float*)    alloc((size_t)MROWS * NDBL * 4);
  float*     dtv    = (float*)    alloc((size_t)MROWS * DIN * 4);
  float*     yv     = (float*)    alloc((size_t)MROWS * DIN * 4);
  _Float16*  yg16   = (_Float16*) alloc((size_t)MROWS * DPAD * 2);
  float*     yo     = (float*)    alloc((size_t)MROWS * CCH * 4);
  float*     x2     = (float*)    alloc((size_t)LQ * CCH * 4);
  float*     xn2    = (float*)    alloc((size_t)LQ * CCH * 4);
  float*     h1     = (float*)    alloc((size_t)NCC * LQ * 4);
  float*     h2     = (float*)    alloc((size_t)CCH * LQ * 4);
  float*     attn   = (float*)    alloc(128 * 4);
  float*     aP     = (float*)    alloc((size_t)NDIR * DIN * NCHK * SST * 4);
  float*     bAgg   = (float*)    alloc((size_t)NDIR * DIN * NCHK * SST * 4);
  float*     hin    = (float*)    alloc((size_t)NDIR * DIN * NCHK * SST * 4);
  _Float16*  w16in  = (_Float16*) alloc((size_t)NPROJ * CCH * 2);
  _Float16*  w16xp  = (_Float16*) alloc((size_t)NDBL * DPAD * 2);
  _Float16*  w16out = (_Float16*) alloc((size_t)CCH * DPAD * 2);

  const int NB = 2;
  for (int i = 0; i < NB; ++i) {
    int s = i % 2;
    const float* xin  = (i == 0) ? x_in : xbuf;
    float*       xout = (i == NB - 1) ? (float*)d_out : xbuf;
    const int*   iids = inv_ids + (size_t)s * 4 * LQ;

    // ---- stage weights to f16 (zero-padded K) ----
    k_wcvt<<<(NPROJ * CCH + 255) / 256, 256, 0, stream>>>(
        in_w + (size_t)i * NPROJ * CCH, w16in, NPROJ, NPROJ, CCH, CCH);
    k_wcvt<<<(NDBL * DPAD + 255) / 256, 256, 0, stream>>>(
        xp_w + (size_t)i * 38 * DIN, w16xp, NDBL, 38, DPAD, DIN);
    k_wcvt<<<(CCH * DPAD + 255) / 256, 256, 0, stream>>>(
        out_w + (size_t)i * CCH * DIN, w16out, CCH, CCH, DPAD, DIN);

    // ---- LN1 + scatter into 4 scan orders ----
    k_ln1_gather<<<LQ / 256, 256, 0, stream>>>(
        xin, ln1_g + i * CCH, ln1_b + i * CCH, iids, xn, xs16);

    // ---- in_proj: (16384 x 288) = xs16 (K=96) @ W^T ----
    k_gemm_wmma<<<(MROWS / 16) * (NPROJ / 16) / 8, 256, 0, stream>>>(
        xs16, w16in, xz, MROWS / 16, NPROJ / 16, CCH, CCH, NPROJ, CCH);

    // ---- depthwise causal conv + SiLU ----
    k_conv_silu<<<MROWS * DPAD / 256, 256, 0, stream>>>(
        xz, conv_w + (size_t)i * DIN * 3, conv_b + i * DIN, xm, xm16);

    // ---- x_proj: (16384 x 48) = xm16 (K=160) @ W^T ----
    k_gemm_wmma<<<(MROWS / 16) * (NDBL / 16) / 8, 256, 0, stream>>>(
        xm16, w16xp, dbl, MROWS / 16, NDBL / 16, DPAD, DPAD, NDBL, DPAD);

    // ---- dt = softplus(proj) ----
    k_dt<<<MROWS * DIN / 256, 256, 0, stream>>>(
        dbl, dt_w + (size_t)i * DIN * RR, dt_b + i * DIN, dtv);

    // ---- chunked selective scan (3 phases) ----
    k_scan_p1<<<(NDIR * DIN * NCHK) / 16, 256, 0, stream>>>(
        dtv, xm, dbl, a_log + (size_t)i * DIN * SST, aP, bAgg);
    k_scan_p2<<<(NDIR * DIN * SST) / 256, 256, 0, stream>>>(aP, bAgg, hin);
    k_scan_p3<<<(NDIR * DIN * NCHK) / 16, 256, 0, stream>>>(
        dtv, xm, dbl, a_log + (size_t)i * DIN * SST, Dp + i * DIN, hin, yv);

    // ---- gate ----
    k_gate<<<MROWS * DPAD / 256, 256, 0, stream>>>(yv, xz, yg16);

    // ---- out_proj: (16384 x 96) = yg16 (K=160) @ W^T ----
    k_gemm_wmma<<<(MROWS / 16) * (CCH / 16) / 8, 256, 0, stream>>>(
        yg16, w16out, yo, MROWS / 16, CCH / 16, DPAD, DPAD, CCH, DPAD);

    // ---- inverse-gather merge + skip ----
    k_merge<<<LQ * CCH / 256, 256, 0, stream>>>(yo, xn, iids, ss + i * CCH, x2);

    // ---- LN2 ----
    k_ln<<<LQ / 256, 256, 0, stream>>>(x2, ln2_g + i * CCH, ln2_b + i * CCH, xn2);

    // ---- CAB ----
    k_conv1<<<NCC * LQ / 256, 256, 0, stream>>>(
        xn2, c1w + (size_t)i * NCC * CCH * 9, c1b + i * NCC, h1);
    k_conv2<<<CCH * LQ / 256, 256, 0, stream>>>(
        h1, c2w + (size_t)i * CCH * NCC * 9, c2b + i * CCH, h2);
    k_attn<<<1, 96, 0, stream>>>(
        h2, a1w + (size_t)i * 3 * CCH, a1b + i * 3,
        a2w + (size_t)i * CCH * 3, a2b + i * CCH, attn);

    // ---- residual combine ----
    k_combine<<<LQ * CCH / 256, 256, 0, stream>>>(x2, h2, attn, ss2 + i * CCH, xout);
  }
}
